// Memory_55095840473137
// MI455X (gfx1250) — matrix operand order repaired and checked by reading, not compiled, not run on hardware
//
#include <hip/hip_runtime.h>
#include <hip/hip_bf16.h>

// Problem sizes (fixed by the reference).
#define B_SZ   8192
#define MEM_SZ 2048
#define DK_SZ  64
#define DV_SZ  64
#define KCHUNK 256            // K-blocking depth for the big masked product
#define DT_LD  (2 * B_SZ)     // leading dim of padded dT (zeros interleaved)

typedef __attribute__((ext_vector_type(16))) __bf16 v16bf;
typedef __attribute__((ext_vector_type(8)))  __bf16 v8bf;
typedef __attribute__((ext_vector_type(4)))  __bf16 v4bf;
typedef __attribute__((ext_vector_type(8)))  float  v8f;

static __device__ inline v8f vzero8() {
    v8f z = {0.f,0.f,0.f,0.f,0.f,0.f,0.f,0.f};
    return z;
}

static __device__ inline v8f wmma_bf16(v16bf a, v16bf b, v8f c) {
    // D = A(16x32) * B(32x16) + C, f32 accumulate.
    return __builtin_amdgcn_wmma_f32_16x16x32_bf16(false, a, false, b, (short)0, c,
                                                   false, false);
}

// ---- CDNA5 async global->LDS copy (16B per lane), ASYNCcnt-tracked ----------
static __device__ inline void async_copy16(void* lds_dst, const void* gsrc) {
    // Flat LDS addresses carry the byte offset in the low 32 bits.
    unsigned ldsoff = (unsigned)(unsigned long long)(uintptr_t)lds_dst;
    asm volatile("global_load_async_to_lds_b128 %0, %1, off"
                 :: "v"(ldsoff), "v"(gsrc) : "memory");
}
static __device__ inline void wait_async0() {
    asm volatile("s_wait_asynccnt 0x0" ::: "memory");
}
static __device__ inline void wait_ds0() {
    asm volatile("s_wait_dscnt 0x0" ::: "memory");
}

// ---- Operand loaders (CDNA5 wave32 WMMA register layouts) -------------------
// A operand (M x K = 16x32): lane holds row (row0 + lane%16) of row-major src.
//   halves h -> K = kbase + 16*g + h (contiguous)
static __device__ inline v16bf ld_A_bf16(const __bf16* src, int ld, int row0,
                                         int kbase, int lane) {
    const __bf16* p = src + (size_t)(row0 + (lane & 15)) * ld + kbase
                          + ((lane >> 4) & 1) * 16;
    v8bf lo = *(const v8bf*)p;
    v8bf hi = *(const v8bf*)(p + 8);
    return __builtin_shufflevector(lo, hi, 0,1,2,3,4,5,6,7,8,9,10,11,12,13,14,15);
}

// Same, converting from f32 with a folded scale factor.
static __device__ inline v16bf ld_A_f32(const float* src, int ld, int row0,
                                        int kbase, int lane, float scale) {
    const float* p = src + (size_t)(row0 + (lane & 15)) * ld + kbase
                         + ((lane >> 4) & 1) * 16;
    v16bf o;
    #pragma unroll
    for (int h = 0; h < 8; ++h) o[h] = (__bf16)(p[h] * scale);
    #pragma unroll
    for (int h = 0; h < 8; ++h) o[h + 8] = (__bf16)(p[h + 8] * scale);
    return o;
}

// B operand (K x N = 32x16): lane holds column (col0 + lane%16); its 16 halves
// are 16 CONTIGUOUS K values starting at kbase + 16*g (src is K-major per
// column, i.e. rows of the transposed matrix) -> one aligned 32B load.
static __device__ inline v16bf ld_B_bf16(const __bf16* src, int ld, int col0,
                                         int kbase, int lane) {
    const __bf16* p = src + (size_t)(col0 + (lane & 15)) * ld + kbase
                          + ((lane >> 4) & 1) * 16;
    return *(const v16bf*)p;
}

// ---- K0: K -> bf16 row-major, V -> bf16 transposed (64 x 2048) --------------
__global__ void ttt_prep(const float* __restrict__ Kst, const float* __restrict__ Vst,
                         __bf16* __restrict__ Kbf, __bf16* __restrict__ Vt) {
    int idx = blockIdx.x * blockDim.x + threadIdx.x;
    if (idx < MEM_SZ * DK_SZ) {
        Kbf[idx] = (__bf16)Kst[idx];
        int mrow = idx >> 6, c = idx & 63;
        Vt[(size_t)c * MEM_SZ + mrow] = (__bf16)Vst[idx];
    }
}

// ---- K1: a_q / a_k = softmax(x @ K^T * scale), stored bf16 ------------------
// 512 threads (16 waves) per 16-row tile; scale folded into bf16 conversion.
__global__ __launch_bounds__(512) void ttt_softmax(const float* __restrict__ q,
                                                   const float* __restrict__ k,
                                                   const __bf16* __restrict__ Kbf,
                                                   __bf16* __restrict__ aq,
                                                   __bf16* __restrict__ ak) {
    const int lane = threadIdx.x & 31;
    const int wave = threadIdx.x >> 5;
    const int g = (lane >> 4) & 1;
    const int n = lane & 15;
    const int ntiles = B_SZ / 16;                       // 512
    const bool isQ = (int)blockIdx.x < ntiles;
    const int row0 = ((int)blockIdx.x % ntiles) * 16;
    const float* x = isQ ? q : k;
    __bf16* out = isQ ? aq : ak;

    v16bf A0 = ld_A_f32(x, DK_SZ, row0, 0, lane, 0.125f);   // 1/sqrt(64) folded
    v16bf A1 = ld_A_f32(x, DK_SZ, row0, 32, lane, 0.125f);

    v8f acc[8];
    #pragma unroll
    for (int ct = 0; ct < 8; ++ct) {
        int col0 = wave * 128 + ct * 16;
        v8f c = vzero8();
        c = wmma_bf16(A0, ld_B_bf16(Kbf, DK_SZ, col0, 0, lane), c);
        c = wmma_bf16(A1, ld_B_bf16(Kbf, DK_SZ, col0, 32, lane), c);
        acc[ct] = c;
    }

    __shared__ float red[16][16];
    __shared__ float rowstat[16];

    float m[8];
    #pragma unroll
    for (int r = 0; r < 8; ++r) {
        float v0 = acc[0][r];
        #pragma unroll
        for (int ct = 1; ct < 8; ++ct) v0 = fmaxf(v0, acc[ct][r]);
        m[r] = v0;
    }
    #pragma unroll
    for (int off = 1; off < 16; off <<= 1)
        #pragma unroll
        for (int r = 0; r < 8; ++r) m[r] = fmaxf(m[r], __shfl_xor(m[r], off, 32));
    if (n == 0) {
        #pragma unroll
        for (int r = 0; r < 8; ++r) red[wave][8 * g + r] = m[r];
    }
    __syncthreads();
    if (threadIdx.x < 16) {
        float v0 = red[0][threadIdx.x];
        #pragma unroll
        for (int w = 1; w < 16; ++w) v0 = fmaxf(v0, red[w][threadIdx.x]);
        rowstat[threadIdx.x] = v0;
    }
    __syncthreads();
    float rmax[8];
    #pragma unroll
    for (int r = 0; r < 8; ++r) rmax[r] = rowstat[8 * g + r];
    __syncthreads();

    float s[8];
    #pragma unroll
    for (int r = 0; r < 8; ++r) {
        float t = 0.f;
        #pragma unroll
        for (int ct = 0; ct < 8; ++ct) {
            float e = __expf(acc[ct][r] - rmax[r]);
            acc[ct][r] = e;
            t += e;
        }
        s[r] = t;
    }
    #pragma unroll
    for (int off = 1; off < 16; off <<= 1)
        #pragma unroll
        for (int r = 0; r < 8; ++r) s[r] += __shfl_xor(s[r], off, 32);
    if (n == 0) {
        #pragma unroll
        for (int r = 0; r < 8; ++r) red[wave][8 * g + r] = s[r];
    }
    __syncthreads();
    if (threadIdx.x < 16) {
        float v0 = 0.f;
        #pragma unroll
        for (int w = 0; w < 16; ++w) v0 += red[w][threadIdx.x];
        rowstat[threadIdx.x] = v0;
    }
    __syncthreads();

    #pragma unroll
    for (int r = 0; r < 8; ++r) {
        float inv = 1.0f / rowstat[8 * g + r];
        size_t base = (size_t)(row0 + 8 * g + r) * MEM_SZ + wave * 128 + n;
        #pragma unroll
        for (int ct = 0; ct < 8; ++ct)
            out[base + ct * 16] = (__bf16)(acc[ct][r] * inv);
    }
}

// ---- K2: d = a_k @ V - v, stored transposed + K-padded bf16 -----------------
// Layout: dTp[col][jt*32 + (row%16)] = d[row][col], entries +16..31 zeroed so
// K3 gets its zero-padded B operand and K4 skips the gaps -- no cndmasks.
__global__ __launch_bounds__(128) void ttt_d(const __bf16* __restrict__ ak,
                                             const __bf16* __restrict__ Vt,
                                             const float* __restrict__ v,
                                             __bf16* __restrict__ dTp) {
    const int lane = threadIdx.x & 31;
    const int wave = threadIdx.x >> 5;                  // 4 waves = 4 DV tiles
    const int g = (lane >> 4) & 1;
    const int n = lane & 15;
    const int row0 = (int)blockIdx.x * 16;

    v8f acc = vzero8();
    for (int kc = 0; kc < MEM_SZ; kc += 32)
        acc = wmma_bf16(ld_A_bf16(ak, MEM_SZ, row0, kc, lane),
                        ld_B_bf16(Vt, MEM_SZ, wave * 16, kc, lane), acc);

    const int col = wave * 16 + n;
    const size_t base = (size_t)col * DT_LD + (size_t)blockIdx.x * 32;
    #pragma unroll
    for (int r = 0; r < 8; ++r) {
        int row = row0 + 8 * g + r;
        float dv = acc[r] - v[(size_t)row * DV_SZ + col];
        dTp[base + 8 * g + r] = (__bf16)dv;
        dTp[base + 16 + 8 * g + r] = (__bf16)0.f;       // zero pad group
    }
}

// ---- K3a: r = a_q @ V (initializes r before the masked-sum atomics) ---------
__global__ __launch_bounds__(128) void ttt_rinit(const __bf16* __restrict__ aq,
                                                 const __bf16* __restrict__ Vt,
                                                 float* __restrict__ r_out) {
    const int lane = threadIdx.x & 31;
    const int wave = threadIdx.x >> 5;
    const int g = (lane >> 4) & 1;
    const int n = lane & 15;
    const int row0 = (int)blockIdx.x * 16;

    v8f acc = vzero8();
    for (int kc = 0; kc < MEM_SZ; kc += 32)
        acc = wmma_bf16(ld_A_bf16(aq, MEM_SZ, row0, kc, lane),
                        ld_B_bf16(Vt, MEM_SZ, wave * 16, kc, lane), acc);

    const int col = wave * 16 + n;
    #pragma unroll
    for (int r = 0; r < 8; ++r)
        r_out[(size_t)(row0 + 8 * g + r) * DV_SZ + col] = acc[r];
}

// ---- K3b: r -= tril(a_q @ a_k^T, -1) @ d, 128x128 tile-pair units -----------
// Unit (i, jb): rows [i*128, i*128+128) x cols [jb*128, jb*128+128), jb <= i.
// Both operand panels staged in LDS per 256-wide K chunk via async-to-LDS DMA;
// wave w owns row tile w (8 at-accumulators), then masks, re-layouts through
// per-wave LDS, multiplies by padded dT and atomically accumulates into r.
__global__ __launch_bounds__(256) void ttt_rtiles(const __bf16* __restrict__ aq,
                                                  const __bf16* __restrict__ ak,
                                                  const __bf16* __restrict__ dTp,
                                                  float* __restrict__ r_out) {
    const int lane = threadIdx.x & 31;
    const int wave = threadIdx.x >> 5;                  // 8 waves = 8 row tiles
    const int g = (lane >> 4) & 1;
    const int n = lane & 15;

    // blockIdx -> (i, jb) in the lower-triangular unit grid.
    int idx = (int)blockIdx.x;
    int i = (int)((sqrtf(8.0f * (float)idx + 1.0f) - 1.0f) * 0.5f);
    while ((i + 1) * (i + 2) / 2 <= idx) ++i;
    while (i * (i + 1) / 2 > idx) --i;
    const int jb = idx - i * (i + 1) / 2;

    const int rbase = i * 128;
    const int cbase = jb * 128;

    __shared__ __attribute__((aligned(32))) __bf16 Ast[128 * KCHUNK];
    __shared__ __attribute__((aligned(32))) __bf16 Bst[128 * KCHUNK];
    __shared__ float atile[8][256];

    v8f at[8];
    #pragma unroll
    for (int jl = 0; jl < 8; ++jl) at[jl] = vzero8();

    for (int kc = 0; kc < MEM_SZ; kc += KCHUNK) {
        __syncthreads();
        wait_ds0();                                      // LDS reads drained
        {   // Stage A (a_q rows) and B (a_k rows): 16B per async op.
            int t = (int)threadIdx.x;
            #pragma unroll
            for (int j = 0; j < 16; ++j) {
                int s = t + 256 * j;                     // 16B slot 0..4095
                int e = s * 8;                           // element offset
                int row = e >> 8;                        // / KCHUNK
                int col = e & (KCHUNK - 1);
                async_copy16(&Ast[e], aq + (size_t)(rbase + row) * MEM_SZ + kc + col);
                async_copy16(&Bst[e], ak + (size_t)(cbase + row) * MEM_SZ + kc + col);
            }
        }
        wait_async0();
        __syncthreads();

        #pragma unroll
        for (int kk = 0; kk < KCHUNK; kk += 32) {
            const __bf16* pa = &Ast[(wave * 16 + n) * KCHUNK + kk + 16 * g];
            v8bf lo = *(const v8bf*)pa;
            v8bf hi = *(const v8bf*)(pa + 8);
            v16bf Aop = __builtin_shufflevector(lo, hi,
                        0,1,2,3,4,5,6,7,8,9,10,11,12,13,14,15);
            #pragma unroll
            for (int jl = 0; jl < 8; ++jl) {
                const __bf16* pb = &Bst[(jl * 16 + n) * KCHUNK + kk + 16 * g];
                v16bf Bop = *(const v16bf*)pb;
                at[jl] = wmma_bf16(Aop, Bop, at[jl]);
            }
        }
    }

    // Epilogue: mask, re-layout (negated, K-padded), multiply by d, accumulate.
    const int ittile = i * 8 + wave;
    const __bf16 zb = (__bf16)0.0f;
    v8f acc[4];
    #pragma unroll
    for (int ct = 0; ct < 4; ++ct) acc[ct] = vzero8();

    for (int jl = 0; jl < 8; ++jl) {
        const int jttile = jb * 8 + jl;
        if (jttile > ittile) break;                     // wave-uniform
        #pragma unroll
        for (int r = 0; r < 8; ++r) {
            int M = 8 * g + r;
            float val = -at[jl][r];
            if (jttile == ittile && n >= M) val = 0.f;  // strict lower triangle
            atile[wave][M * 16 + n] = val;
        }
        v16bf Aop;
        #pragma unroll
        for (int h = 0; h < 16; ++h)
            Aop[h] = g ? zb : (__bf16)atile[wave][n * 16 + h];

        #pragma unroll
        for (int ct = 0; ct < 4; ++ct) {
            const __bf16* pb = dTp + (size_t)(ct * 16 + n) * DT_LD
                                   + jttile * 32 + 16 * g;   // g=1 reads zeros
            v16bf Bop = *(const v16bf*)pb;
            acc[ct] = wmma_bf16(Aop, Bop, acc[ct]);     // accumulates -a @ d
        }
    }

    #pragma unroll
    for (int ct = 0; ct < 4; ++ct)
        #pragma unroll
        for (int r = 0; r < 8; ++r) {
            int row = rbase + wave * 16 + 8 * g + r;
            atomicAdd(&r_out[(size_t)row * DV_SZ + ct * 16 + n], acc[ct][r]);
        }
}

// ---- K4: V_new = V - a_k^T @ d ---------------------------------------------
__global__ __launch_bounds__(128) void ttt_vnew(const __bf16* __restrict__ ak,
                                                const __bf16* __restrict__ dTp,
                                                const float* __restrict__ V,
                                                float* __restrict__ vnew) {
    const int lane = threadIdx.x & 31;
    const int wave = threadIdx.x >> 5;                  // 4 waves = 4 DV tiles
    const int g = (lane >> 4) & 1;
    const int n = lane & 15;
    const int m0 = (int)blockIdx.x * 16;

    __shared__ __attribute__((aligned(32))) __bf16 stage[32 * 16];
    v8f acc = vzero8();

    for (int bc = 0; bc < B_SZ; bc += 32) {
        __syncthreads();
        {   // coalesced stage: ak[bc..bc+31][m0..m0+15]
            int e = (int)threadIdx.x * 4;               // 512 elems, 4/thread
            int krow = e >> 4, mcol = e & 15;
            *(v4bf*)&stage[e] =
                *(const v4bf*)(ak + (size_t)(bc + krow) * MEM_SZ + m0 + mcol);
        }
        __syncthreads();
        v16bf Aop;                                      // rows = memory slots
        #pragma unroll
        for (int h = 0; h < 8; ++h) Aop[h] = stage[(16 * g + h) * 16 + n];
        #pragma unroll
        for (int h = 0; h < 8; ++h) Aop[8 + h] = stage[(16 * g + 8 + h) * 16 + n];
        // B from padded dT, skipping the zero gaps: rows bc+16g .. bc+16g+15.
        const __bf16* pb = dTp + (size_t)(wave * 16 + n) * DT_LD
                               + ((bc >> 4) + g) * 32;
        acc = wmma_bf16(Aop, *(const v16bf*)pb, acc);
    }

    const int col = wave * 16 + n;
    #pragma unroll
    for (int r = 0; r < 8; ++r) {
        int mrow = m0 + 8 * g + r;
        vnew[(size_t)mrow * DV_SZ + col] =
            V[(size_t)mrow * DV_SZ + col] - acc[r];
    }
}

extern "C" void kernel_launch(void* const* d_in, const int* in_sizes, int n_in,
                              void* d_out, int out_size, void* d_ws, size_t ws_size,
                              hipStream_t stream) {
    (void)in_sizes; (void)n_in; (void)out_size; (void)ws_size;
    const float* q   = (const float*)d_in[0];
    const float* k   = (const float*)d_in[1];
    const float* v   = (const float*)d_in[2];
    const float* Kst = (const float*)d_in[3];
    const float* Vst = (const float*)d_in[4];

    float* r_out = (float*)d_out;                       // (B, DV)
    float* vnew  = r_out + (size_t)B_SZ * DV_SZ;        // (MEM, DV)

    // Workspace: Kbf 256KB | Vt 256KB | aq 32MB | ak 32MB | dTp 2MB  (~66.5MB)
    char* ws = (char*)d_ws;
    __bf16* Kbf = (__bf16*)(ws);
    __bf16* Vt  = (__bf16*)(ws + (size_t)256 * 1024);
    __bf16* aq  = (__bf16*)(ws + (size_t)512 * 1024);
    __bf16* ak  = (__bf16*)(ws + (size_t)512 * 1024 + (size_t)B_SZ * MEM_SZ * 2);
    __bf16* dTp = (__bf16*)(ws + (size_t)512 * 1024 + (size_t)2 * B_SZ * MEM_SZ * 2);

    const int NBR = B_SZ / 128;                         // 64 row blocks
    const int NUNITS = NBR * (NBR + 1) / 2;             // 2080 tile-pair units

    ttt_prep<<<(MEM_SZ * DK_SZ + 255) / 256, 256, 0, stream>>>(Kst, Vst, Kbf, Vt);
    ttt_softmax<<<2 * (B_SZ / 16), 512, 0, stream>>>(q, k, Kbf, aq, ak);
    ttt_d<<<B_SZ / 16, 128, 0, stream>>>(ak, Vt, v, dTp);
    ttt_rinit<<<B_SZ / 16, 128, 0, stream>>>(aq, Vt, r_out);
    ttt_rtiles<<<NUNITS, 256, 0, stream>>>(aq, ak, dTp, r_out);
    ttt_vnew<<<MEM_SZ / 16, 128, 0, stream>>>(ak, dTp, Vst, vnew);
}
